// Seq2SeqWAttention_71837622993506
// MI455X (gfx1250) — matrix operand order repaired and checked by reading, not compiled, not run on hardware
//
#include <hip/hip_runtime.h>
#include <math.h>

typedef __attribute__((ext_vector_type(16))) __bf16 v16bf;
typedef __attribute__((ext_vector_type(8)))  float  v8f;

namespace {
constexpr int Vv = 30611, Ee = 1024, Hh = 128, H2 = 256, Bb = 16, Ss = 400, Tt = 100;
constexpr int DIN = Ee + H2;   // 1280 decoder input
constexpr int GD  = 4 * H2;    // 1024 decoder gate width
constexpr int GE  = 4 * Hh;    // 512 encoder gate width
constexpr int NBLKV = 240, VROWS = 128;
constexpr int START_TOKEN = 101;
}

// ---------------- WMMA helpers (gfx1250 wave32, 16x16x32 bf16) ----------------

__device__ __forceinline__ v8f wmma_bf16(v16bf a, v16bf b, v8f c) {
  return __builtin_amdgcn_wmma_f32_16x16x32_bf16(false, a, false, b, (short)0, c,
                                                 false, false);
}

__device__ __forceinline__ v8f zero8() {
  v8f z;
#pragma unroll
  for (int i = 0; i < 8; ++i) z[i] = 0.f;
  return z;
}

// A fragment: 16x32 bf16, row m = lane&15, K half = lane>>4, 16 contiguous K per lane
__device__ __forceinline__ v16bf load_a_bf(const __bf16* base, int ld, int m0, int k0) {
  int lane = threadIdx.x & 31;
  int m = lane & 15, kh = lane >> 4;
  const __bf16* r = base + (size_t)(m0 + m) * ld + k0 + kh * 16;
  v16bf a;
#pragma unroll
  for (int i = 0; i < 8; ++i) { a[i] = r[i]; a[8 + i] = r[8 + i]; }
  return a;
}

// A fragment loaded from f32 with on-the-fly bf16 conversion
__device__ __forceinline__ v16bf load_a_f32(const float* base, int ld, int m0, int k0) {
  int lane = threadIdx.x & 31;
  int m = lane & 15, kh = lane >> 4;
  const float* r = base + (size_t)(m0 + m) * ld + k0 + kh * 16;
  v16bf a;
#pragma unroll
  for (int i = 0; i < 8; ++i) { a[i] = (__bf16)r[i]; a[8 + i] = (__bf16)r[8 + i]; }
  return a;
}

// B fragment: logical 32x16, B[k][n] = W[n*ldk + k] (row-major weight [N][K]);
// lane n = lane&15, K half = lane>>4, 16 contiguous K per lane.
__device__ __forceinline__ v16bf load_b_bf(const __bf16* w, int ldk, int n0, int k0) {
  int lane = threadIdx.x & 31;
  int n = lane & 15, kh = lane >> 4;
  const __bf16* r = w + (size_t)(n0 + n) * ldk + k0 + kh * 16;
  v16bf b;
#pragma unroll
  for (int i = 0; i < 16; ++i) b[i] = r[i];
  return b;
}

__device__ __forceinline__ v16bf load_b_bf_clamp(const __bf16* w, int ldk, int n0,
                                                 int k0, int nmax) {
  int lane = threadIdx.x & 31;
  int n = lane & 15, kh = lane >> 4;
  int row = n0 + n; if (row >= nmax) row = nmax - 1;
  const __bf16* r = w + (size_t)row * ldk + k0 + kh * 16;
  v16bf b;
#pragma unroll
  for (int i = 0; i < 16; ++i) b[i] = r[i];
  return b;
}

__device__ __forceinline__ float sigf(float x) { return 1.f / (1.f + expf(-x)); }

// ---------------- prep kernels ----------------

__global__ void k_cvt(const float* __restrict__ s, __bf16* __restrict__ d, int n) {
  for (int i = blockIdx.x * blockDim.x + threadIdx.x; i < n; i += gridDim.x * blockDim.x)
    d[i] = (__bf16)s[i];
}

__global__ void k_gather_x(const float* __restrict__ emb, const int* __restrict__ article,
                           __bf16* __restrict__ X) {
  int n = Ss * Bb * Ee;
  for (int i = blockIdx.x * blockDim.x + threadIdx.x; i < n; i += gridDim.x * blockDim.x) {
    int e = i % Ee; int sb = i / Ee; int b = sb % Bb; int s = sb / Bb;
    int tok = article[b * Ss + s];
    X[i] = (__bf16)emb[(size_t)tok * Ee + e];
  }
}

__global__ void k_init(const float* __restrict__ emb, float* __restrict__ dec_in,
                       float* __restrict__ accum) {
  int n = Bb * DIN;
  for (int i = blockIdx.x * blockDim.x + threadIdx.x; i < n; i += gridDim.x * blockDim.x) {
    int k = i % DIN;
    dec_in[i] = (k < Ee) ? emb[(size_t)START_TOKEN * Ee + k] : 0.f;
  }
  if (blockIdx.x == 0 && threadIdx.x < 2) accum[threadIdx.x] = 0.f;
}

// ---------------- encoder input GEMM: xg = X @ Wih^T + b  (both dirs) ----------------

__global__ void k_enc_gemm(const __bf16* __restrict__ X,
                           const __bf16* __restrict__ Wf_, const __bf16* __restrict__ Wb_,
                           const float* __restrict__ bf_, const float* __restrict__ bb_,
                           float* __restrict__ xg) {
  int s = blockIdx.x;
  int dir = blockIdx.z;
  int wave = threadIdx.x >> 5;
  int nt = blockIdx.y * 8 + wave;                 // 0..31 over N=512
  const __bf16* W = dir ? Wb_ : Wf_;
  const float* bias = dir ? bb_ : bf_;
  float* out = xg + (size_t)dir * Ss * Bb * GE;
  v8f acc = zero8();
  for (int k0 = 0; k0 < Ee; k0 += 32) {
    v16bf a = load_a_bf(X, Ee, s * 16, k0);       // rows = (s, b=0..15)
    v16bf b = load_b_bf(W, Ee, nt * 16, k0);
    acc = wmma_bf16(a, b, acc);
  }
  int lane = threadIdx.x & 31, n = lane & 15, kh = lane >> 4;
  float bn = bias[nt * 16 + n];
#pragma unroll
  for (int v = 0; v < 8; ++v) {
    int m = v + 8 * kh;                           // batch b
    out[((size_t)s * Bb + m) * GE + nt * 16 + n] = acc[v] + bn;
  }
}

// ---------------- encoder recurrence (persistent, 1 block/direction) ----------------
// block 256 = 8 waves; wave w owns columns j = 16w..16w+15 of all four gates.
// All 16 Whh B-fragments (4 gates x 4 K-steps = 128 VGPRs) are hoisted into
// registers before the 400-step loop: steady state has NO global weight loads
// on the serial critical path. Ping-pong h buffers: one barrier per step.

__global__ void __launch_bounds__(256) k_enc_rec(
    const float* __restrict__ xg,
    const __bf16* __restrict__ Whhf, const __bf16* __restrict__ Whhb,
    float* __restrict__ enc_out, __bf16* __restrict__ enc_out_b,
    float* __restrict__ dec_h, float* __restrict__ dec_c) {
  __shared__ __bf16 hb[2][Bb * Hh];
  int dir = blockIdx.x;
  const __bf16* Whh = dir ? Whhb : Whhf;
  const float* xgd = xg + (size_t)dir * Ss * Bb * GE;
  int wave = threadIdx.x >> 5, lane = threadIdx.x & 31;
  int n = lane & 15, kh = lane >> 4, j0 = wave * 16, j = j0 + n;

  v16bf Bfr[16];                       // [gate*4 + kstep], register-resident
#pragma unroll
  for (int g = 0; g < 4; ++g)
#pragma unroll
    for (int kk = 0; kk < 4; ++kk)
      Bfr[g * 4 + kk] = load_b_bf(Whh, Hh, g * Hh + j0, kk * 32);

  for (int i = threadIdx.x; i < Bb * Hh; i += blockDim.x) hb[0][i] = (__bf16)0.f;
  float creg[8];
#pragma unroll
  for (int v = 0; v < 8; ++v) creg[v] = 0.f;
  __syncthreads();
  int p = 0;
  for (int step = 0; step < Ss; ++step) {
    int sio = dir ? (Ss - 1 - step) : step;
    v8f gi = zero8(), gf = zero8(), gg = zero8(), go = zero8();
#pragma unroll
    for (int kk = 0; kk < 4; ++kk) {
      v16bf A = load_a_bf(hb[p], Hh, 0, kk * 32);
      gi = wmma_bf16(A, Bfr[0 * 4 + kk], gi);
      gf = wmma_bf16(A, Bfr[1 * 4 + kk], gf);
      gg = wmma_bf16(A, Bfr[2 * 4 + kk], gg);
      go = wmma_bf16(A, Bfr[3 * 4 + kk], go);
    }
    const float* xr = xgd + (size_t)sio * Bb * GE;
#pragma unroll
    for (int v = 0; v < 8; ++v) {
      int b = v + 8 * kh;
      float i_ = gi[v] + xr[b * GE + 0 * Hh + j];
      float f_ = gf[v] + xr[b * GE + 1 * Hh + j];
      float g_ = gg[v] + xr[b * GE + 2 * Hh + j];
      float o_ = go[v] + xr[b * GE + 3 * Hh + j];
      float cn = sigf(f_) * creg[v] + sigf(i_) * tanhf(g_);
      creg[v] = cn;
      float hn = sigf(o_) * tanhf(cn);
      hb[p ^ 1][b * Hh + j] = (__bf16)hn;   // write other buffer: no pre-barrier
      size_t ro = ((size_t)b * Ss + sio) * H2 + dir * Hh + j;
      enc_out[ro] = hn;
      enc_out_b[ro] = (__bf16)hn;
      if (step == Ss - 1) {
        dec_h[b * H2 + dir * Hh + j] = hn;
        dec_c[b * H2 + dir * Hh + j] = cn;
      }
    }
    __syncthreads();
    p ^= 1;
  }
}

// ---------------- enc_ws = enc_out @ Ws^T + bs ----------------

__global__ void k_enc_ws(const __bf16* __restrict__ enc_out_b, const __bf16* __restrict__ Wsb,
                         const float* __restrict__ bs, float* __restrict__ enc_ws) {
  int mt = blockIdx.x;              // 400 tiles x 16 rows = 6400 rows (b*Ss+s)
  int wave = threadIdx.x >> 5;      // 0..3 over N=64
  v8f acc = zero8();
  for (int k0 = 0; k0 < H2; k0 += 32) {
    v16bf a = load_a_bf(enc_out_b, H2, mt * 16, k0);
    v16bf b = load_b_bf(Wsb, H2, wave * 16, k0);
    acc = wmma_bf16(a, b, acc);
  }
  int lane = threadIdx.x & 31, n = lane & 15, kh = lane >> 4;
  float bn = bs[wave * 16 + n];
#pragma unroll
  for (int v = 0; v < 8; ++v) {
    int m = v + 8 * kh;
    enc_ws[(size_t)(mt * 16 + m) * 64 + wave * 16 + n] = acc[v] + bn;
  }
}

// ---------------- decoder LSTM cell (serial part only; 1 block, 16 waves) ----------------

__global__ void __launch_bounds__(512) k_dec_lstm(
    const float* __restrict__ dec_in, float* __restrict__ dec_h, float* __restrict__ dec_c,
    __bf16* __restrict__ h_bf,
    const __bf16* __restrict__ Wihd, const __bf16* __restrict__ Whhd,
    const float* __restrict__ b_d) {
  int wave = threadIdx.x >> 5, lane = threadIdx.x & 31;
  int n = lane & 15, kh = lane >> 4;
  int j0 = wave * 16, j = j0 + n;

  // wave w owns i/f/g/o columns 16w..16w+15: nonlinearity on fragments in regs
  v8f gi = zero8(), gf = zero8(), gg = zero8(), go = zero8();
  for (int k0 = 0; k0 < DIN; k0 += 32) {
    v16bf A = load_a_f32(dec_in, DIN, 0, k0);
    gi = wmma_bf16(A, load_b_bf(Wihd, DIN, 0 * H2 + j0, k0), gi);
    gf = wmma_bf16(A, load_b_bf(Wihd, DIN, 1 * H2 + j0, k0), gf);
    gg = wmma_bf16(A, load_b_bf(Wihd, DIN, 2 * H2 + j0, k0), gg);
    go = wmma_bf16(A, load_b_bf(Wihd, DIN, 3 * H2 + j0, k0), go);
  }
  for (int k0 = 0; k0 < H2; k0 += 32) {
    v16bf A = load_a_f32(dec_h, H2, 0, k0);
    gi = wmma_bf16(A, load_b_bf(Whhd, H2, 0 * H2 + j0, k0), gi);
    gf = wmma_bf16(A, load_b_bf(Whhd, H2, 1 * H2 + j0, k0), gf);
    gg = wmma_bf16(A, load_b_bf(Whhd, H2, 2 * H2 + j0, k0), gg);
    go = wmma_bf16(A, load_b_bf(Whhd, H2, 3 * H2 + j0, k0), go);
  }
  float hreg[8], creg[8];
#pragma unroll
  for (int v = 0; v < 8; ++v) {
    int b = v + 8 * kh;
    float i_ = gi[v] + b_d[0 * H2 + j];
    float f_ = gf[v] + b_d[1 * H2 + j];
    float g_ = gg[v] + b_d[2 * H2 + j];
    float o_ = go[v] + b_d[3 * H2 + j];
    float cold = dec_c[b * H2 + j];
    creg[v] = sigf(f_) * cold + sigf(i_) * tanhf(g_);
    hreg[v] = sigf(o_) * tanhf(creg[v]);
  }
  __syncthreads();   // all waves done reading dec_h as WMMA A operand
#pragma unroll
  for (int v = 0; v < 8; ++v) {
    int b = v + 8 * kh;
    dec_c[b * H2 + j] = creg[v];
    dec_h[b * H2 + j] = hreg[v];
    h_bf[b * H2 + j] = (__bf16)hreg[v];
  }
}

// ---------------- attention + context + NLL reduce (parallel; 1 block per batch row) ----

__global__ void __launch_bounds__(256) k_attn_ctx(
    int t, const float* __restrict__ dec_h, float* __restrict__ dec_in,
    const float* __restrict__ Wh, const float* __restrict__ bh,
    const float* __restrict__ Wf, const float* __restrict__ bfv,
    const float* __restrict__ enc_ws, const float* __restrict__ enc_out,
    const int* __restrict__ alen, const float* __restrict__ emb,
    const int* __restrict__ abstract,
    const float* __restrict__ partials, const __bf16* __restrict__ h_bf,
    const __bf16* __restrict__ Wpb, const float* __restrict__ bp,
    float* __restrict__ accum) {
  __shared__ float a_l[64];
  __shared__ float wf_l[64];
  __shared__ float sc[Ss];
  __shared__ float red[8];
  int b = blockIdx.x;
  int tid = threadIdx.x;
  int wave = tid >> 5, lane = tid & 31;

  if (tid < 64) {
    wf_l[tid] = Wf[tid];
    float acc = bh[tid];
    const float* hr = &dec_h[b * H2];
    const float* wr = &Wh[tid * H2];
    for (int k = 0; k < H2; ++k) acc += hr[k] * wr[k];
    a_l[tid] = acc;
  }
  __syncthreads();

  // additive attention scores + length mask
  float bf0 = bfv[0];
  int len = alen[b];
  for (int s = tid; s < Ss; s += 256) {
    const float* er = &enc_ws[((size_t)b * Ss + s) * 64];
    float v = bf0;
    for (int k = 0; k < 64; ++k) v += tanhf(er[k] + a_l[k]) * wf_l[k];
    sc[s] = fminf(v, (s < len) ? 9999.f : -9999.f);
  }
  __syncthreads();

  // block softmax over S=400 (normalization folded into ctx accumulate)
  float m = -1e30f;
  for (int s = tid; s < Ss; s += 256) m = fmaxf(m, sc[s]);
  for (int off = 16; off; off >>= 1) m = fmaxf(m, __shfl_xor(m, off, 32));
  if (lane == 0) red[wave] = m;
  __syncthreads();
  m = red[0];
  for (int w = 1; w < 8; ++w) m = fmaxf(m, red[w]);
  float sum = 0.f;
  for (int s = tid; s < Ss; s += 256) {
    float e = expf(sc[s] - m);
    sc[s] = e;
    sum += e;
  }
  for (int off = 16; off; off >>= 1) sum += __shfl_xor(sum, off, 32);
  __syncthreads();
  if (lane == 0) red[wave] = sum;
  __syncthreads();
  sum = 0.f;
  for (int w = 0; w < 8; ++w) sum += red[w];
  float inv = 1.f / sum;

  // ctx: coalesced streaming of this row's enc_out slice (one j per thread)
  {
    int j = tid;
    float acc = 0.f;
    const float* ep = &enc_out[((size_t)b * Ss) * H2 + j];
    for (int s = 0; s < Ss; ++s) acc += sc[s] * ep[(size_t)s * H2];
    dec_in[b * DIN + Ee + j] = acc * inv;
  }
  // teacher-forcing embedding for next step input
  int tok = abstract[b * Tt + t];
  for (int e = tid; e < Ee; e += 256) dec_in[b * DIN + e] = emb[(size_t)tok * Ee + e];

  // NLL for (b, t): combine the 240 vocab partials
  if (wave == 0) {
    float mm = -1e30f;
    for (int p = lane; p < NBLKV; p += 32) mm = fmaxf(mm, partials[(p * Bb + b) * 2]);
    for (int off = 16; off; off >>= 1) mm = fmaxf(mm, __shfl_xor(mm, off, 32));
    float ss = 0.f;
    for (int p = lane; p < NBLKV; p += 32)
      ss += partials[(p * Bb + b) * 2 + 1] * expf(partials[(p * Bb + b) * 2] - mm);
    for (int off = 16; off; off >>= 1) ss += __shfl_xor(ss, off, 32);
    if (lane == 0) {
      float lse = mm + logf(ss);
      float dot = bp[tok];
      for (int k = 0; k < H2; ++k)
        dot += (float)h_bf[b * H2 + k] * (float)Wpb[(size_t)tok * H2 + k];
      if (tok != 0) {
        atomicAdd(&accum[0], lse - dot);
        atomicAdd(&accum[1], 1.f);
      }
    }
  }
}

// ---------------- vocab projection partials (logits never materialized) ----------------

__global__ void __launch_bounds__(256) k_vocab(const __bf16* __restrict__ h_bf,
                                               const __bf16* __restrict__ Wpb,
                                               const float* __restrict__ bp,
                                               float* __restrict__ partials) {
  __shared__ float lg[Bb][VROWS];
  __shared__ float red[Bb][16];
  int wave = threadIdx.x >> 5, lane = threadIdx.x & 31;
  int n = lane & 15, kh = lane >> 4;
  int n0 = blockIdx.x * VROWS + wave * 16;
  v8f acc = zero8();
  for (int k0 = 0; k0 < H2; k0 += 32) {
    v16bf a = load_a_bf(h_bf, H2, 0, k0);
    v16bf b = load_b_bf_clamp(Wpb, H2, n0, k0, Vv);
    acc = wmma_bf16(a, b, acc);
  }
  int col = n0 + n;
  bool ok = col < Vv;
  float bias = ok ? bp[col] : 0.f;
  int lcol = wave * 16 + n;
#pragma unroll
  for (int v = 0; v < 8; ++v) {
    int b = v + 8 * kh;
    lg[b][lcol] = ok ? (acc[v] + bias) : -1e30f;
  }
  __syncthreads();
  // parallel (max, sum-exp) reduce: 16 threads per batch row, 8 columns each
  int rb = threadIdx.x & 15, chunk = threadIdx.x >> 4;
  {
    const float* row = &lg[rb][chunk * 8];
    float m = row[0];
#pragma unroll
    for (int c = 1; c < 8; ++c) m = fmaxf(m, row[c]);
    red[rb][chunk] = m;
  }
  __syncthreads();
  float gm = red[rb][0];
#pragma unroll
  for (int c = 1; c < 16; ++c) gm = fmaxf(gm, red[rb][c]);
  __syncthreads();
  {
    const float* row = &lg[rb][chunk * 8];
    float s = 0.f;
#pragma unroll
    for (int c = 0; c < 8; ++c) s += expf(row[c] - gm);
    red[rb][chunk] = s;
  }
  __syncthreads();
  if (threadIdx.x < Bb) {
    int b = threadIdx.x;     // == rb for these threads
    float s = 0.f;
#pragma unroll
    for (int c = 0; c < 16; ++c) s += red[b][c];
    partials[(blockIdx.x * Bb + b) * 2 + 0] = gm;
    partials[(blockIdx.x * Bb + b) * 2 + 1] = s;
  }
}

// ---------------- final scalar ----------------

__global__ void k_finish(const float* __restrict__ accum, float* __restrict__ out) {
  if (threadIdx.x == 0 && blockIdx.x == 0)
    out[0] = accum[0] / fmaxf(accum[1], 1.f);
}

// ---------------- host orchestration ----------------

extern "C" void kernel_launch(void* const* d_in, const int* in_sizes, int n_in,
                              void* d_out, int out_size, void* d_ws, size_t ws_size,
                              hipStream_t stream) {
  (void)in_sizes; (void)n_in; (void)out_size; (void)ws_size;
  const float* emb      = (const float*)d_in[0];
  const int*   article  = (const int*)d_in[1];
  const int*   alen     = (const int*)d_in[2];
  const int*   abstract = (const int*)d_in[3];
  const float* Wih_ef = (const float*)d_in[4];
  const float* Whh_ef = (const float*)d_in[5];
  const float* b_ef   = (const float*)d_in[6];
  const float* Wih_eb = (const float*)d_in[7];
  const float* Whh_eb = (const float*)d_in[8];
  const float* b_eb   = (const float*)d_in[9];
  const float* Wih_d  = (const float*)d_in[10];
  const float* Whh_d  = (const float*)d_in[11];
  const float* b_d    = (const float*)d_in[12];
  const float* Wp     = (const float*)d_in[13];
  const float* bp     = (const float*)d_in[14];
  const float* Ws     = (const float*)d_in[15];
  const float* bs     = (const float*)d_in[16];
  const float* Wh     = (const float*)d_in[17];
  const float* bh     = (const float*)d_in[18];
  const float* Wf     = (const float*)d_in[19];
  const float* bf_    = (const float*)d_in[20];
  float* out = (float*)d_out;

  char* base = (char*)d_ws;
  size_t off = 0;
  auto take = [&](size_t bytes) -> char* {
    off = (off + 255) & ~(size_t)255;
    char* p = base + off;
    off += bytes;
    return p;
  };
  __bf16* Wih_ef_b = (__bf16*)take((size_t)GE * Ee * 2);
  __bf16* Whh_ef_b = (__bf16*)take((size_t)GE * Hh * 2);
  __bf16* Wih_eb_b = (__bf16*)take((size_t)GE * Ee * 2);
  __bf16* Whh_eb_b = (__bf16*)take((size_t)GE * Hh * 2);
  __bf16* Wih_d_b  = (__bf16*)take((size_t)GD * DIN * 2);
  __bf16* Whh_d_b  = (__bf16*)take((size_t)GD * H2 * 2);
  __bf16* Wp_b     = (__bf16*)take((size_t)Vv * H2 * 2);
  __bf16* Ws_b     = (__bf16*)take((size_t)64 * H2 * 2);
  __bf16* Xb       = (__bf16*)take((size_t)Ss * Bb * Ee * 2);
  float*  xg       = (float*)take((size_t)2 * Ss * Bb * GE * 4);
  float*  enc_out  = (float*)take((size_t)Bb * Ss * H2 * 4);
  __bf16* enc_out_b= (__bf16*)take((size_t)Bb * Ss * H2 * 2);
  float*  enc_ws   = (float*)take((size_t)Bb * Ss * 64 * 4);
  float*  dec_h    = (float*)take((size_t)Bb * H2 * 4);
  float*  dec_c    = (float*)take((size_t)Bb * H2 * 4);
  float*  dec_in   = (float*)take((size_t)Bb * DIN * 4);
  __bf16* h_bf     = (__bf16*)take((size_t)Bb * H2 * 2);
  float*  partials = (float*)take((size_t)NBLKV * Bb * 2 * 4);
  float*  accum    = (float*)take(16 * 4);

  auto cgrid = [](int n) { int b = (n + 255) / 256; return b > 2048 ? 2048 : b; };
  k_cvt<<<cgrid(GE * Ee), 256, 0, stream>>>(Wih_ef, Wih_ef_b, GE * Ee);
  k_cvt<<<cgrid(GE * Hh), 256, 0, stream>>>(Whh_ef, Whh_ef_b, GE * Hh);
  k_cvt<<<cgrid(GE * Ee), 256, 0, stream>>>(Wih_eb, Wih_eb_b, GE * Ee);
  k_cvt<<<cgrid(GE * Hh), 256, 0, stream>>>(Whh_eb, Whh_eb_b, GE * Hh);
  k_cvt<<<cgrid(GD * DIN), 256, 0, stream>>>(Wih_d, Wih_d_b, GD * DIN);
  k_cvt<<<cgrid(GD * H2), 256, 0, stream>>>(Whh_d, Whh_d_b, GD * H2);
  k_cvt<<<cgrid(Vv * H2), 256, 0, stream>>>(Wp, Wp_b, Vv * H2);
  k_cvt<<<cgrid(64 * H2), 256, 0, stream>>>(Ws, Ws_b, 64 * H2);
  k_gather_x<<<2048, 256, 0, stream>>>(emb, article, Xb);
  k_init<<<(Bb * DIN + 255) / 256, 256, 0, stream>>>(emb, dec_in, accum);

  k_enc_gemm<<<dim3(Ss, 4, 2), 256, 0, stream>>>(Xb, Wih_ef_b, Wih_eb_b, b_ef, b_eb, xg);
  k_enc_rec<<<2, 256, 0, stream>>>(xg, Whh_ef_b, Whh_eb_b, enc_out, enc_out_b, dec_h, dec_c);
  k_enc_ws<<<Ss, 128, 0, stream>>>(enc_out_b, Ws_b, bs, enc_ws);

  for (int t = 0; t < Tt; ++t) {
    k_dec_lstm<<<1, 512, 0, stream>>>(dec_in, dec_h, dec_c, h_bf,
                                      Wih_d_b, Whh_d_b, b_d);
    k_vocab<<<NBLKV, 256, 0, stream>>>(h_bf, Wp_b, bp, partials);
    k_attn_ctx<<<Bb, 256, 0, stream>>>(t, dec_h, dec_in, Wh, bh, Wf, bf_,
                                       enc_ws, enc_out, alen, emb, abstract,
                                       partials, h_bf, Wp_b, bp, accum);
  }
  k_finish<<<1, 64, 0, stream>>>(accum, out);
}
